// OptimizedGraphTransGeoGCN_78546361909451
// MI455X (gfx1250) — matrix operand order repaired and checked by reading, not compiled, not run on hardware
//
#include <hip/hip_runtime.h>

// ---------------------------------------------------------------------------
// Types for CDNA5 WMMA
// ---------------------------------------------------------------------------
typedef __attribute__((ext_vector_type(16))) __bf16 v16bf;
typedef __attribute__((ext_vector_type(8)))  float  v8f;

#define TM 128
#define TN 64
#define LDA_S 40   // A tile row stride: 32 k + 8 pad (bf16 elements)
#define LDB_T 40   // transposed B tile row stride: 32 k + 8 pad

// ---------------------------------------------------------------------------
// GEMM: C[M,Nout] = A[M,256] @ B[256,Nout], optionally scaled per-row by
// rowscale[m] in the epilogue.  bf16 inputs (converted on LDS stage),
// f32 accumulation via v_wmma_f32_16x16x32_bf16.
// Block = 256 threads = 8 waves; block tile 128x64; wave tile 32x32 (2x2 WMMA).
// B is staged TRANSPOSED in LDS so fragment loads are K-contiguous b128s.
// ---------------------------------------------------------------------------
__global__ __launch_bounds__(256) void ggcn_gemm_wmma(
    const float* __restrict__ A, const float* __restrict__ B,
    float* __restrict__ C, const float* __restrict__ rowscale,
    int M, int Nout, int useScale)
{
    __shared__ __bf16 sA[TM * LDA_S];
    __shared__ __bf16 sBt[TN * LDB_T];   // [n][k]

    const int t    = threadIdx.x;
    const int lane = t & 31;
    const int wave = t >> 5;
    const int wm   = wave & 3;   // 0..3  (M direction)
    const int wn   = wave >> 2;  // 0..1  (N direction)
    const int row0 = blockIdx.x * TM;
    const int n0   = blockIdx.y * TN;

    const int l16 = lane & 15;
    const int hi  = lane >> 4;   // 0 or 1

    const bool fullTile = (row0 + TM) <= M;

    v8f acc[2][2] = {};

    for (int k0 = 0; k0 < 256; k0 += 32) {
        // ---- stage A tile: 128 rows x 32 k, f32 -> bf16 ----
        {
            const int r0 = t >> 3;           // 0..31
            const int c4 = (t & 7) << 2;     // 0,4,...,28
            if (fullTile) {
                #pragma unroll
                for (int i = 0; i < 4; ++i) {
                    const int r = r0 + i * 32;
                    float4 v = *reinterpret_cast<const float4*>(
                        A + (size_t)(row0 + r) * 256 + k0 + c4);
                    __bf16* d = &sA[r * LDA_S + c4];
                    d[0] = (__bf16)v.x; d[1] = (__bf16)v.y;
                    d[2] = (__bf16)v.z; d[3] = (__bf16)v.w;
                }
            } else {
                #pragma unroll
                for (int i = 0; i < 4; ++i) {
                    const int r = r0 + i * 32;
                    float4 v = make_float4(0.f, 0.f, 0.f, 0.f);
                    if (row0 + r < M)
                        v = *reinterpret_cast<const float4*>(
                            A + (size_t)(row0 + r) * 256 + k0 + c4);
                    __bf16* d = &sA[r * LDA_S + c4];
                    d[0] = (__bf16)v.x; d[1] = (__bf16)v.y;
                    d[2] = (__bf16)v.z; d[3] = (__bf16)v.w;
                }
            }
        }
        // ---- stage B tile transposed: 32 k x 64 n -> sBt[n][k] ----
        {
            const int r0 = t >> 4;           // k offset 0..15
            const int c4 = (t & 15) << 2;    // n offset 0..60
            #pragma unroll
            for (int i = 0; i < 2; ++i) {
                const int r = r0 + i * 16;
                float4 v = *reinterpret_cast<const float4*>(
                    B + (size_t)(k0 + r) * Nout + n0 + c4);
                sBt[(c4 + 0) * LDB_T + r] = (__bf16)v.x;
                sBt[(c4 + 1) * LDB_T + r] = (__bf16)v.y;
                sBt[(c4 + 2) * LDB_T + r] = (__bf16)v.z;
                sBt[(c4 + 3) * LDB_T + r] = (__bf16)v.w;
            }
        }
        __syncthreads();

        // ---- build fragments (ISA 16-bit A 16x32 / B 32x16 layouts) ----
        v16bf afrag[2], bfrag[2];
        #pragma unroll
        for (int i = 0; i < 2; ++i) {
            const int r  = wm * 32 + i * 16 + l16;
            const int kb = hi * 8;
            #pragma unroll
            for (int j = 0; j < 8; ++j) afrag[i][j]     = sA[r * LDA_S + kb + j];
            #pragma unroll
            for (int j = 0; j < 8; ++j) afrag[i][8 + j] = sA[r * LDA_S + 16 + kb + j];
        }
        #pragma unroll
        for (int jn = 0; jn < 2; ++jn) {
            const int n  = wn * 32 + jn * 16 + l16;
            const int kb = hi * 8;
            #pragma unroll
            for (int j = 0; j < 8; ++j) bfrag[jn][j]     = sBt[n * LDB_T + kb + j];
            #pragma unroll
            for (int j = 0; j < 8; ++j) bfrag[jn][8 + j] = sBt[n * LDB_T + 16 + kb + j];
        }

        #pragma unroll
        for (int i = 0; i < 2; ++i)
            #pragma unroll
            for (int j = 0; j < 2; ++j)
                acc[i][j] = __builtin_amdgcn_wmma_f32_16x16x32_bf16(
                    false, afrag[i], false, bfrag[j], (short)0, acc[i][j], false, false);
        __syncthreads();
    }

    // ---- epilogue store (C/D layout: VGPR r -> row = base + hi*8 + r) ----
    #pragma unroll
    for (int i = 0; i < 2; ++i) {
        const int rbase = row0 + wm * 32 + i * 16 + hi * 8;
        #pragma unroll
        for (int j = 0; j < 2; ++j) {
            const int n = n0 + wn * 32 + j * 16 + l16;
            #pragma unroll
            for (int r = 0; r < 8; ++r) {
                const int m = rbase + r;
                if (m < M) {
                    const float s = useScale ? rowscale[m] : 1.0f;
                    C[(size_t)m * Nout + n] = acc[i][j][r] * s;
                }
            }
        }
    }
}

// ---------------------------------------------------------------------------
// Helper kernels
// ---------------------------------------------------------------------------
__global__ void ggcn_fill(float* p, float v, int n)
{
    int i = blockIdx.x * 256 + threadIdx.x;
    if (i < n) p[i] = v;
}

// Column-wise sum / sumsq over X[N,256] (thread t owns column t).
__global__ __launch_bounds__(256) void ggcn_col_stats(
    const float* __restrict__ X, float* colsum, float* colsq, int N)
{
    const int t = threadIdx.x;
    float s = 0.f, q = 0.f;
    const int total  = N * 256;
    const int stride = gridDim.x * 256;
    for (int idx = blockIdx.x * 256 + t; idx < total; idx += stride) {
        const float v = X[idx];
        s += v; q += v * v;
    }
    atomicAdd(&colsum[t], s);
    atomicAdd(&colsq[t],  q);
}

__global__ void ggcn_bn_finalize(const float* colsum, const float* colsq,
                                 const float* g, const float* b,
                                 float* Ac, float* Cc, float invN)
{
    const int t = threadIdx.x;
    const float m   = colsum[t] * invN;
    const float var = colsq[t] * invN - m * m;
    const float a   = g[t] * rsqrtf(var + 1e-5f);
    Ac[t] = a;
    Cc[t] = b[t] - m * a;
}

// Y = X*A[c]+C[c] (+ Y if residual)
__global__ void ggcn_bn_apply(const float* __restrict__ X,
                              const float* __restrict__ Ac,
                              const float* __restrict__ Cc,
                              float* __restrict__ Y, int N, int residual)
{
    const int idx = blockIdx.x * 256 + threadIdx.x;
    if (idx >= N * 256) return;
    const int c = threadIdx.x;
    float v = X[idx] * Ac[c] + Cc[c];
    if (residual) v += Y[idx];
    Y[idx] = v;
}

__global__ void ggcn_deg(const int* __restrict__ dst, float* deg, int E)
{
    int e = blockIdx.x * 256 + threadIdx.x;
    if (e < E) atomicAdd(&deg[dst[e]], 1.f);
}

__global__ void ggcn_dinv(const float* deg, float* dinv, int N)
{
    int n = blockIdx.x * 256 + threadIdx.x;
    if (n < N) dinv[n] = rsqrtf(deg[n] + 1.f);
}

// agg[dst] += hs[src]  (dinv already folded into hs; dinv[dst] applied later)
__global__ void ggcn_scatter(const float* __restrict__ hs,
                             const int* __restrict__ src,
                             const int* __restrict__ dst,
                             float* __restrict__ agg, int E)
{
    const int idx = blockIdx.x * 256 + threadIdx.x;
    const int e = idx >> 8, c = idx & 255;
    if (e < E)
        atomicAdd(&agg[(size_t)dst[e] * 256 + c], hs[(size_t)src[e] * 256 + c]);
}

// y = relu(dinv[n]*(agg+hs)+bias[c]); fused BN stats accumulation.
__global__ __launch_bounds__(256) void ggcn_epilogue(
    const float* __restrict__ agg, const float* __restrict__ hs,
    const float* __restrict__ dinv, const float* __restrict__ bias,
    float* __restrict__ y, float* colsum, float* colsq, int N)
{
    const int t = threadIdx.x;
    const float bb = bias[t];
    float s = 0.f, q = 0.f;
    const int total  = N * 256;
    const int stride = gridDim.x * 256;
    for (int idx = blockIdx.x * 256 + t; idx < total; idx += stride) {
        const int n = idx >> 8;
        float v = dinv[n] * (agg[idx] + hs[idx]) + bb;
        v = fmaxf(v, 0.f);
        y[idx] = v;
        s += v; q += v * v;
    }
    atomicAdd(&colsum[t], s);
    atomicAdd(&colsq[t],  q);
}

// score[n] = leaky_relu(s1[n,:]+ba1, .01) . Wa2 + ba2   (one wave per node)
__global__ __launch_bounds__(256) void ggcn_score(
    const float* __restrict__ s1, const float* __restrict__ ba1,
    const float* __restrict__ Wa2, const float* __restrict__ ba2,
    float* __restrict__ score, int N)
{
    const int lane = threadIdx.x & 31;
    const int node = blockIdx.x * 8 + (threadIdx.x >> 5);
    if (node >= N) return;
    float acc = 0.f;
    #pragma unroll
    for (int i = 0; i < 4; ++i) {
        const int c = lane + i * 32;
        float v = s1[(size_t)node * 128 + c] + ba1[c];
        v = v > 0.f ? v : 0.01f * v;
        acc += v * Wa2[c];
    }
    #pragma unroll
    for (int off = 16; off > 0; off >>= 1)
        acc += __shfl_xor(acc, off, 32);
    if (lane == 0) score[node] = acc + ba2[0];
}

__global__ __launch_bounds__(256) void ggcn_red_max(const float* x, float* red, int N)
{
    __shared__ float sm[256];
    float m = -3.4e38f;
    for (int i = blockIdx.x * 256 + threadIdx.x; i < N; i += gridDim.x * 256)
        m = fmaxf(m, x[i]);
    sm[threadIdx.x] = m; __syncthreads();
    for (int s = 128; s > 0; s >>= 1) {
        if (threadIdx.x < s) sm[threadIdx.x] = fmaxf(sm[threadIdx.x], sm[threadIdx.x + s]);
        __syncthreads();
    }
    if (threadIdx.x == 0) red[blockIdx.x] = sm[0];
}

__global__ __launch_bounds__(256) void ggcn_final_max(const float* red, float* out, int nb)
{
    __shared__ float sm[256];
    float m = -3.4e38f;
    for (int i = threadIdx.x; i < nb; i += 256) m = fmaxf(m, red[i]);
    sm[threadIdx.x] = m; __syncthreads();
    for (int s = 128; s > 0; s >>= 1) {
        if (threadIdx.x < s) sm[threadIdx.x] = fmaxf(sm[threadIdx.x], sm[threadIdx.x + s]);
        __syncthreads();
    }
    if (threadIdx.x == 0) out[0] = sm[0];
}

__global__ __launch_bounds__(256) void ggcn_exp_sum(const float* score, const float* smax,
                                                    float* red, int N)
{
    __shared__ float sm[256];
    const float m = smax[0];
    float s = 0.f;
    for (int i = blockIdx.x * 256 + threadIdx.x; i < N; i += gridDim.x * 256)
        s += expf(score[i] - m);
    sm[threadIdx.x] = s; __syncthreads();
    for (int st = 128; st > 0; st >>= 1) {
        if (threadIdx.x < st) sm[threadIdx.x] += sm[threadIdx.x + st];
        __syncthreads();
    }
    if (threadIdx.x == 0) red[blockIdx.x] = sm[0];
}

__global__ __launch_bounds__(256) void ggcn_final_sum(const float* red, float* out, int nb)
{
    __shared__ float sm[256];
    float s = 0.f;
    for (int i = threadIdx.x; i < nb; i += 256) s += red[i];
    sm[threadIdx.x] = s; __syncthreads();
    for (int st = 128; st > 0; st >>= 1) {
        if (threadIdx.x < st) sm[threadIdx.x] += sm[threadIdx.x + st];
        __syncthreads();
    }
    if (threadIdx.x == 0) out[0] = sm[0];
}

__global__ void ggcn_cnt(const int* __restrict__ batch, float* cnt, int N)
{
    int n = blockIdx.x * 256 + threadIdx.x;
    if (n < N) atomicAdd(&cnt[batch[n]], 1.f);
}

__global__ void ggcn_pool(const float* __restrict__ xp, const float* __restrict__ score,
                          const float* __restrict__ smax, const float* __restrict__ ssum,
                          const int* __restrict__ batch, float* __restrict__ pooled, int N)
{
    const int idx = blockIdx.x * 256 + threadIdx.x;
    if (idx >= N * 256) return;
    const int n = idx >> 8, c = idx & 255;
    const float w = expf(score[n] - smax[0]) / ssum[0];
    atomicAdd(&pooled[(size_t)batch[n] * 256 + c], xp[idx] * w);
}

__global__ __launch_bounds__(256) void ggcn_out(const float* __restrict__ pooled,
                                                const float* __restrict__ cnt,
                                                const float* __restrict__ Wo,
                                                const float* __restrict__ bo,
                                                float* __restrict__ out)
{
    __shared__ float s0[256], s1[256];
    const int g = blockIdx.x, t = threadIdx.x;
    const float c = fmaxf(cnt[g], 1.f);
    const float p = pooled[(size_t)g * 256 + t] / c;
    s0[t] = p * Wo[t * 2];
    s1[t] = p * Wo[t * 2 + 1];
    __syncthreads();
    for (int s = 128; s > 0; s >>= 1) {
        if (t < s) { s0[t] += s0[t + s]; s1[t] += s1[t + s]; }
        __syncthreads();
    }
    if (t == 0) {
        out[g * 2]     = s0[0] + bo[0];
        out[g * 2 + 1] = s1[0] + bo[1];
    }
}

// ---------------------------------------------------------------------------
// Host launcher
// ---------------------------------------------------------------------------
extern "C" void kernel_launch(void* const* d_in, const int* in_sizes, int n_in,
                              void* d_out, int out_size, void* d_ws, size_t ws_size,
                              hipStream_t stream)
{
    const float* x    = (const float*)d_in[0];
    const int*   ei   = (const int*)d_in[1];
    const int*   batch= (const int*)d_in[2];
    const float* bing = (const float*)d_in[3];
    const float* binb = (const float*)d_in[4];
    const float* W[3]    = {(const float*)d_in[5],  (const float*)d_in[9],  (const float*)d_in[13]};
    const float* bvec[3] = {(const float*)d_in[6],  (const float*)d_in[10], (const float*)d_in[14]};
    const float* bng[3]  = {(const float*)d_in[7],  (const float*)d_in[11], (const float*)d_in[15]};
    const float* bnb[3]  = {(const float*)d_in[8],  (const float*)d_in[12], (const float*)d_in[16]};
    const float* Wa1 = (const float*)d_in[17];
    const float* ba1 = (const float*)d_in[18];
    const float* Wa2 = (const float*)d_in[19];
    const float* ba2 = (const float*)d_in[20];
    const float* Wo  = (const float*)d_in[21];
    const float* bo  = (const float*)d_in[22];
    float* out = (float*)d_out;

    const int N = in_sizes[2];          // 50000
    const int E = in_sizes[1] / 2;      // 800000
    const int G = out_size / 2;         // 64
    const int NF = N * 256;

    const int* src = ei;
    const int* dst = ei + E;

    // Workspace carve-up (floats)
    float* ws     = (float*)d_ws;
    float* xcur   = ws;                 // [N,256] activations
    float* hs     = xcur + NF;          // [N,256] (X@W)*dinv ; also s1 [N,128]
    float* agg    = hs + NF;            // [N,256] scatter target
    float* ybuf   = agg + NF;           // [N,256] relu output
    float* dinv   = ybuf + NF;          // [N]
    float* degb   = dinv + N;           // [N]
    float* score  = degb + N;           // [N]
    float* colsum = score + N;          // [256]
    float* colsq  = colsum + 256;       // [256]
    float* bnA    = colsq + 256;        // [256]
    float* bnC    = bnA + 256;          // [256]
    float* red    = bnC + 256;          // [256]
    float* smax   = red + 256;          // [1]
    float* ssum   = smax + 1;           // [1]
    float* pooled = ssum + 1;           // [G,256]
    float* cnt    = pooled + (size_t)G * 256;  // [G]

    const int elemBlocks = (NF + 255) / 256;   // N blocks
    const float invN = 1.0f / (float)N;

    // ---- input BN ----
    ggcn_fill<<<2, 256, 0, stream>>>(colsum, 0.f, 512);
    ggcn_col_stats<<<1024, 256, 0, stream>>>(x, colsum, colsq, N);
    ggcn_bn_finalize<<<1, 256, 0, stream>>>(colsum, colsq, bing, binb, bnA, bnC, invN);
    ggcn_bn_apply<<<elemBlocks, 256, 0, stream>>>(x, bnA, bnC, xcur, N, 0);

    // ---- degree / dinv (once) ----
    ggcn_fill<<<(N + 255) / 256, 256, 0, stream>>>(degb, 0.f, N);
    ggcn_deg<<<(E + 255) / 256, 256, 0, stream>>>(dst, degb, E);
    ggcn_dinv<<<(N + 255) / 256, 256, 0, stream>>>(degb, dinv, N);

    // ---- 3 GCN layers ----
    dim3 ggrid((N + TM - 1) / TM, 256 / TN);
    for (int l = 0; l < 3; ++l) {
        ggcn_gemm_wmma<<<ggrid, 256, 0, stream>>>(xcur, W[l], hs, dinv, N, 256, 1);
        ggcn_fill<<<elemBlocks, 256, 0, stream>>>(agg, 0.f, NF);
        ggcn_scatter<<<E, 256, 0, stream>>>(hs, src, dst, agg, E);
        ggcn_fill<<<2, 256, 0, stream>>>(colsum, 0.f, 512);
        ggcn_epilogue<<<1024, 256, 0, stream>>>(agg, hs, dinv, bvec[l], ybuf, colsum, colsq, N);
        ggcn_bn_finalize<<<1, 256, 0, stream>>>(colsum, colsq, bng[l], bnb[l], bnA, bnC, invN);
        ggcn_bn_apply<<<elemBlocks, 256, 0, stream>>>(ybuf, bnA, bnC, xcur, N, l > 0 ? 1 : 0);
    }

    // ---- attention: s1 = xcur @ Wa1 (N x 128), score, softmax ----
    dim3 agrid((N + TM - 1) / TM, 128 / TN);
    ggcn_gemm_wmma<<<agrid, 256, 0, stream>>>(xcur, Wa1, hs, dinv, N, 128, 0);
    ggcn_score<<<(N + 7) / 8, 256, 0, stream>>>(hs, ba1, Wa2, ba2, score, N);
    ggcn_red_max<<<256, 256, 0, stream>>>(score, red, N);
    ggcn_final_max<<<1, 256, 0, stream>>>(red, smax, 256);
    ggcn_exp_sum<<<256, 256, 0, stream>>>(score, smax, red, N);
    ggcn_final_sum<<<1, 256, 0, stream>>>(red, ssum, 256);

    // ---- pooling + output ----
    ggcn_fill<<<(G * 256 + G + 255) / 256, 256, 0, stream>>>(pooled, 0.f, G * 256 + G);
    ggcn_cnt<<<(N + 255) / 256, 256, 0, stream>>>(batch, cnt, N);
    ggcn_pool<<<elemBlocks, 256, 0, stream>>>(xcur, score, smax, ssum, batch, pooled, N);
    ggcn_out<<<G, 256, 0, stream>>>(pooled, cnt, Wo, bo, out);
}
